// ConvNeXtBlock_50405736185868
// MI455X (gfx1250) — matrix operand order repaired and checked by reading, not compiled, not run on hardware
//
#include <hip/hip_runtime.h>
#include <hip/hip_bf16.h>

#define B_ 16
#define C_ 512
#define L_ 1024
#define H_ 2048
#define M_ (B_*L_)

typedef __attribute__((ext_vector_type(16))) __bf16 v16bf;
typedef __attribute__((ext_vector_type(8)))  float v8f;

union FragBF { unsigned u[8]; v16bf v; };

// Async global->LDS copy, 16B per lane, tracked by ASYNCcnt (gfx1250).
// vdst = per-lane LDS byte offset, vaddr = per-lane 64-bit global address.
#define ASYNC_LD_B128(ldsoff, gptr)                                   \
  asm volatile("global_load_async_to_lds_b128 %0, %1, off"            \
               :: "v"(ldsoff), "v"(gptr) : "memory")

__device__ __forceinline__ void wait_async0() {
#if __has_builtin(__builtin_amdgcn_s_wait_asynccnt)
  __builtin_amdgcn_s_wait_asynccnt(0);
#else
  asm volatile("s_wait_asynccnt 0x0" ::: "memory");
#endif
}

// Low 32 bits of a generic pointer into LDS == LDS byte offset on AMDGPU.
__device__ __forceinline__ unsigned lds_off(const void* p) {
  return (unsigned)(unsigned long long)p;
}

// Split an f32 into bf16 hi + bf16 lo (RNE both), x ~= hi + lo.
__device__ __forceinline__ void split_bf16(float x, unsigned short& hi, unsigned short& lo) {
  unsigned bx = __float_as_uint(x);
  unsigned rh = bx + 0x7FFFu + ((bx >> 16) & 1u);
  hi = (unsigned short)(rh >> 16);
  float fhi = __uint_as_float((unsigned)hi << 16);
  float res = x - fhi;
  unsigned br = __float_as_uint(res);
  unsigned rl = br + 0x7FFFu + ((br >> 16) & 1u);
  lo = (unsigned short)(rl >> 16);
}

// ---------------- weight split: f32 -> bf16 hi/lo ----------------
__global__ __launch_bounds__(256) void split_arr(const float* __restrict__ w,
    unsigned short* __restrict__ hi, unsigned short* __restrict__ lo, int n) {
  int i = blockIdx.x * 256 + threadIdx.x;
  if (i < n) {
    unsigned short h, l;
    split_bf16(w[i], h, l);
    hi[i] = h; lo[i] = l;
  }
}

// ---------------- depthwise conv1d k=7 pad=3 + per-batch sum/sumsq ----------------
__global__ __launch_bounds__(256) void conv_dw(const float* __restrict__ x,
    const float* __restrict__ w, const float* __restrict__ bias,
    float* __restrict__ out, float* __restrict__ sums) {
  __shared__ float row[L_ + 6];
  __shared__ float rsum[8], rsq[8];
  int t = threadIdx.x;
  int blk = blockIdx.x;
  int b = blk / C_, c = blk % C_;
  const float* xr = x + ((size_t)b * C_ + c) * L_;
  for (int i = t; i < L_ + 6; i += 256) {
    int g = i - 3;
    row[i] = (g >= 0 && g < L_) ? xr[g] : 0.0f;
  }
  float wk[7];
#pragma unroll
  for (int j = 0; j < 7; ++j) wk[j] = w[c * 7 + j];
  float bb = bias[c];
  __syncthreads();
  float lsum = 0.f, lsq = 0.f;
  float* orow = out + ((size_t)b * C_ + c) * L_;
#pragma unroll
  for (int k = 0; k < L_ / 256; ++k) {
    int i = t + k * 256;
    float acc = bb;
#pragma unroll
    for (int j = 0; j < 7; ++j) acc += wk[j] * row[i + j];
    orow[i] = acc;
    lsum += acc; lsq += acc * acc;
  }
  for (int o = 16; o > 0; o >>= 1) {
    lsum += __shfl_down(lsum, o, 32);
    lsq  += __shfl_down(lsq,  o, 32);
  }
  int lane = t & 31, wv = t >> 5;
  if (lane == 0) { rsum[wv] = lsum; rsq[wv] = lsq; }
  __syncthreads();
  if (t == 0) {
    float s = 0.f, q = 0.f;
#pragma unroll
    for (int i = 0; i < 8; ++i) { s += rsum[i]; q += rsq[i]; }
    atomicAdd(&sums[2 * b], s);
    atomicAdd(&sums[2 * b + 1], q);
  }
}

// ---------------- LayerNorm (per-batch scalar mu/var) + transpose + bf16 split ----------------
__global__ __launch_bounds__(256) void norm_split(const float* __restrict__ conv,
    const float* __restrict__ sums, const float* __restrict__ gamma,
    const float* __restrict__ beta, unsigned short* __restrict__ nhi,
    unsigned short* __restrict__ nlo) {
  __shared__ float tile[64][65];
  int t = threadIdx.x;
  int c0 = blockIdx.x * 64, l0 = blockIdx.y * 64, b = blockIdx.z;
  const float invN = 1.0f / (float)(C_ * L_);
  float mu = sums[2 * b] * invN;
  float var = sums[2 * b + 1] * invN - mu * mu;
  float rs = rsqrtf(var + 1e-5f);
  int tj = t & 63, ti = t >> 6;
#pragma unroll
  for (int r = 0; r < 16; ++r) {
    int ci = r * 4 + ti;  // c-local, read coalesced along l
    tile[ci][tj] = conv[(((size_t)b * C_) + (c0 + ci)) * L_ + l0 + tj];
  }
  __syncthreads();
#pragma unroll
  for (int r = 0; r < 16; ++r) {
    int li = r * 4 + ti;  // l-local, write coalesced along c
    int cj = tj;
    int l = l0 + li, c = c0 + cj;
    float v = tile[cj][li];
    float nv = (v - mu) * rs * gamma[l * C_ + c] + beta[l * C_ + c];
    unsigned short h16, l16;
    split_bf16(nv, h16, l16);
    size_t m = (size_t)b * L_ + l;
    nhi[m * C_ + c] = h16;
    nlo[m * C_ + c] = l16;
  }
}

// ---------------- split-bf16 WMMA GEMM core, double-buffered async staging ----------------
// C[m,n] = sum_k A[m,k]*B[n,k]  (B given row-major [N,K], i.e. w1/w2 layout)
// Block tile 64(M) x 128(N), 8 waves of 32 -> each wave 32x32 (four 16x16 accums).
// Tiles staged via global_load_async_to_lds_b128 (ASYNCcnt) into ping-pong LDS
// buffers: iteration computes from buf `cur` while DMA fills buf `cur^1`.
// LDS row stride 20 uints (80B: 16B-aligned rows, conflict-free frag gathers).
// EPI==0: +bias, exact GELU, bf16 hi/lo store.  EPI==1: raw f32 store.
#define LDP 20
template<int Kd, int Nd, int EPI>
__global__ __launch_bounds__(256) void gemm_wmma(
    const unsigned* __restrict__ Ahi, const unsigned* __restrict__ Alo,
    const unsigned* __restrict__ Bhi, const unsigned* __restrict__ Blo,
    const float* __restrict__ bias, float* __restrict__ outF,
    unsigned short* __restrict__ outHi, unsigned short* __restrict__ outLo) {
  constexpr int KU = Kd / 2;  // uints (bf16 pairs) per row
  __shared__ unsigned sAh[2][64 * LDP], sAl[2][64 * LDP];
  __shared__ unsigned sBh[2][128 * LDP], sBl[2][128 * LDP];
  int t = threadIdx.x;
  int m0 = blockIdx.y * 64, n0 = blockIdx.x * 128;
  int lane = t & 31, wave = t >> 5;
  int wm = wave >> 2, wn = wave & 3;
  int half = lane >> 4, lr = lane & 15;

  v8f zero = {0.f, 0.f, 0.f, 0.f, 0.f, 0.f, 0.f, 0.f};
  v8f acc[2][2];
#pragma unroll
  for (int i = 0; i < 2; ++i)
#pragma unroll
    for (int j = 0; j < 2; ++j) acc[i][j] = zero;

  // Per-thread staging slots (loop-invariant LDS destinations, both buffers).
  int ar = t >> 2, aq = t & 3;
  int br1 = (t + 256) >> 2;
  unsigned oAh[2], oAl[2], oBh0[2], oBl0[2], oBh1[2], oBl1[2];
#pragma unroll
  for (int u = 0; u < 2; ++u) {
    oAh[u]  = lds_off(&sAh[u][ar * LDP + aq * 4]);
    oAl[u]  = lds_off(&sAl[u][ar * LDP + aq * 4]);
    oBh0[u] = lds_off(&sBh[u][ar * LDP + aq * 4]);
    oBl0[u] = lds_off(&sBl[u][ar * LDP + aq * 4]);
    oBh1[u] = lds_off(&sBh[u][br1 * LDP + aq * 4]);
    oBl1[u] = lds_off(&sBl[u][br1 * LDP + aq * 4]);
  }
  // Per-thread global source pointers (advance by ku uints per step).
  const unsigned* pAh  = Ahi + (size_t)(m0 + ar)  * KU + aq * 4;
  const unsigned* pAl  = Alo + (size_t)(m0 + ar)  * KU + aq * 4;
  const unsigned* pBh0 = Bhi + (size_t)(n0 + ar)  * KU + aq * 4;
  const unsigned* pBl0 = Blo + (size_t)(n0 + ar)  * KU + aq * 4;
  const unsigned* pBh1 = Bhi + (size_t)(n0 + br1) * KU + aq * 4;
  const unsigned* pBl1 = Blo + (size_t)(n0 + br1) * KU + aq * 4;

  auto issue = [&](int buf, int ku) {
    ASYNC_LD_B128(oAh[buf],  pAh  + ku);
    ASYNC_LD_B128(oAl[buf],  pAl  + ku);
    ASYNC_LD_B128(oBh0[buf], pBh0 + ku);
    ASYNC_LD_B128(oBl0[buf], pBl0 + ku);
    ASYNC_LD_B128(oBh1[buf], pBh1 + ku);
    ASYNC_LD_B128(oBl1[buf], pBl1 + ku);
  };

  auto step = [&](int k0, int buf) {
    wait_async0();    // transfers for buf (issued last iteration) have landed
    __syncthreads();  // visible to all waves; prior reads of buf^1 retired
    int kn = k0 + 32;
    if (kn < Kd) issue(buf ^ 1, kn >> 1);  // overlap DMA with WMMA below
    if (k0 + 64 < Kd) {                    // prefetch tile after next
      __builtin_prefetch(pAh  + ((k0 + 64) >> 1), 0, 1);
      __builtin_prefetch(pBh0 + ((k0 + 64) >> 1), 0, 1);
    }
    FragBF ah[2], al2[2], bh[2], bl2[2];
#pragma unroll
    for (int i = 0; i < 2; ++i) {
      // A 16x32 frag: lanes 0-15 -> K0..7 & K16..23; lanes 16-31 -> K8..15 & K24..31
      const unsigned* rp = &sAh[buf][(wm * 32 + i * 16 + lr) * LDP];
      const unsigned* rq = &sAl[buf][(wm * 32 + i * 16 + lr) * LDP];
      int base = half * 4;
#pragma unroll
      for (int j = 0; j < 4; ++j) {
        ah[i].u[j]      = rp[base + j];
        ah[i].u[4 + j]  = rp[8 + base + j];
        al2[i].u[j]     = rq[base + j];
        al2[i].u[4 + j] = rq[8 + base + j];
      }
    }
#pragma unroll
    for (int i = 0; i < 2; ++i) {
      // B 32x16 frag: lanes 0-15 -> K0..7 & K16..23; lanes 16-31 -> K8..15 & K24..31 (N = lane&15)
      const unsigned* rp = &sBh[buf][(wn * 32 + i * 16 + lr) * LDP];
      const unsigned* rq = &sBl[buf][(wn * 32 + i * 16 + lr) * LDP];
      int base = half * 4;
#pragma unroll
      for (int j = 0; j < 4; ++j) {
        bh[i].u[j]      = rp[base + j];
        bh[i].u[4 + j]  = rp[8 + base + j];
        bl2[i].u[j]     = rq[base + j];
        bl2[i].u[4 + j] = rq[8 + base + j];
      }
    }
#pragma unroll
    for (int i = 0; i < 2; ++i)
#pragma unroll
      for (int j = 0; j < 2; ++j) {
        acc[i][j] = __builtin_amdgcn_wmma_f32_16x16x32_bf16(
            false, ah[i].v, false, bh[j].v, (short)0, acc[i][j], false, false);
        acc[i][j] = __builtin_amdgcn_wmma_f32_16x16x32_bf16(
            false, ah[i].v, false, bl2[j].v, (short)0, acc[i][j], false, false);
        acc[i][j] = __builtin_amdgcn_wmma_f32_16x16x32_bf16(
            false, al2[i].v, false, bh[j].v, (short)0, acc[i][j], false, false);
      }
  };

  issue(0, 0);  // prologue: fill buffer 0
  for (int k0 = 0; k0 < Kd; k0 += 64) {  // Kd/32 is even for both GEMMs
    step(k0, 0);
    step(k0 + 32, 1);
  }

  // epilogue (C/D layout: lanes 0-15 M=r, lanes 16-31 M=8+r; N = lane&15)
#pragma unroll
  for (int i = 0; i < 2; ++i)
#pragma unroll
    for (int j = 0; j < 2; ++j)
#pragma unroll
      for (int r = 0; r < 8; ++r) {
        int m = m0 + wm * 32 + i * 16 + half * 8 + r;
        int n = n0 + wn * 32 + j * 16 + lr;
        float v = acc[i][j][r];
        if (EPI == 0) {
          v += bias[n];
          float g = 0.5f * v * (1.0f + erff(v * 0.70710678118654752f));
          unsigned short hh, ll;
          split_bf16(g, hh, ll);
          outHi[(size_t)m * Nd + n] = hh;
          outLo[(size_t)m * Nd + n] = ll;
        } else {
          outF[(size_t)m * Nd + n] = v;
        }
      }
}

// ---------------- final: [M,C] -> [B,C,L] transpose + b2 + residual ----------------
__global__ __launch_bounds__(256) void final_out(const float* __restrict__ yt,
    const float* __restrict__ b2, const float* __restrict__ x,
    float* __restrict__ out) {
  __shared__ float tile[64][65];
  int t = threadIdx.x;
  int c0 = blockIdx.x * 64, l0 = blockIdx.y * 64, b = blockIdx.z;
  int tj = t & 63, ti = t >> 6;
#pragma unroll
  for (int r = 0; r < 16; ++r) {
    int li = r * 4 + ti;  // read coalesced along c
    tile[li][tj] = yt[((size_t)b * L_ + l0 + li) * C_ + c0 + tj];
  }
  __syncthreads();
#pragma unroll
  for (int r = 0; r < 16; ++r) {
    int ci = r * 4 + ti;  // write coalesced along l
    int c = c0 + ci, l = l0 + tj;
    size_t oi = ((size_t)b * C_ + c) * L_ + l;
    out[oi] = tile[tj][ci] + b2[c] + x[oi];
  }
}

extern "C" void kernel_launch(void* const* d_in, const int* in_sizes, int n_in,
                              void* d_out, int out_size, void* d_ws, size_t ws_size,
                              hipStream_t stream) {
  (void)in_sizes; (void)n_in; (void)out_size; (void)ws_size;
  const float* x     = (const float*)d_in[0];
  const float* dw_w  = (const float*)d_in[1];
  const float* dw_b  = (const float*)d_in[2];
  const float* gamma = (const float*)d_in[3];
  const float* beta  = (const float*)d_in[4];
  const float* w1    = (const float*)d_in[5];
  const float* b1    = (const float*)d_in[6];
  const float* w2    = (const float*)d_in[7];
  const float* b2    = (const float*)d_in[8];
  float* out = (float*)d_out;

  char* ws = (char*)d_ws;
  size_t off = 0;
  float* sums = (float*)(ws + off); off += 256;
  float* conv = (float*)(ws + off); off += (size_t)B_ * C_ * L_ * 4;  // aliased as yt later
  unsigned short* nhi = (unsigned short*)(ws + off); off += (size_t)M_ * C_ * 2;
  unsigned short* nlo = (unsigned short*)(ws + off); off += (size_t)M_ * C_ * 2;
  unsigned short* hhi = (unsigned short*)(ws + off); off += (size_t)M_ * H_ * 2;
  unsigned short* hlo = (unsigned short*)(ws + off); off += (size_t)M_ * H_ * 2;
  unsigned short* w1h = (unsigned short*)(ws + off); off += (size_t)H_ * C_ * 2;
  unsigned short* w1l = (unsigned short*)(ws + off); off += (size_t)H_ * C_ * 2;
  unsigned short* w2h = (unsigned short*)(ws + off); off += (size_t)C_ * H_ * 2;
  unsigned short* w2l = (unsigned short*)(ws + off); off += (size_t)C_ * H_ * 2;
  float* yt = conv;  // conv buffer is dead after norm_split; reuse for GEMM2 output

  hipMemsetAsync(sums, 0, 256, stream);
  split_arr<<<(H_ * C_ + 255) / 256, 256, 0, stream>>>(w1, w1h, w1l, H_ * C_);
  split_arr<<<(C_ * H_ + 255) / 256, 256, 0, stream>>>(w2, w2h, w2l, C_ * H_);
  conv_dw<<<B_ * C_, 256, 0, stream>>>(x, dw_w, dw_b, conv, sums);
  norm_split<<<dim3(C_ / 64, L_ / 64, B_), 256, 0, stream>>>(conv, sums, gamma, beta, nhi, nlo);
  gemm_wmma<C_, H_, 0><<<dim3(H_ / 128, M_ / 64), 256, 0, stream>>>(
      (const unsigned*)nhi, (const unsigned*)nlo, (const unsigned*)w1h, (const unsigned*)w1l,
      b1, nullptr, hhi, hlo);
  gemm_wmma<H_, C_, 1><<<dim3(C_ / 128, M_ / 64), 256, 0, stream>>>(
      (const unsigned*)hhi, (const unsigned*)hlo, (const unsigned*)w2h, (const unsigned*)w2l,
      nullptr, yt, nullptr, nullptr);
  final_out<<<dim3(C_ / 64, L_ / 64, B_), 256, 0, stream>>>(yt, b2, x, out);
}
